// AxiomENSORCLN_25838523253157
// MI455X (gfx1250) — matrix operand order repaired and checked by reading, not compiled, not run on hardware
//
#include <hip/hip_runtime.h>
#include <hip/hip_bf16.h>

typedef __attribute__((ext_vector_type(16))) _Float16 v16h;
typedef __attribute__((ext_vector_type(8)))  float    v8f;

#define B_ROWS   262144
#define HIST     24
#define FORE     20
#define HID      256
#define TAU_INT  6

#define WAVES_PER_WG   4
#define ROWS_PER_WAVE  16
#define ROWS_PER_WG    (WAVES_PER_WG * ROWS_PER_WAVE)

// Packed-weight fragment layout: fragment (kt,nt) = 32 lanes x 16 halves,
// base = ((kt*NT + nt)*32 + lane)*16 halves.
#define FRAG_HALVES 512
#define W1_KT 2
#define W1_NT 16
#define W2_KT 8
#define W2_NT 16
#define W3_KT 8
#define W3_NT 16
#define W4_KT 8
#define W4_NT 2
#define W1_OFF 0
#define W2_OFF (W1_OFF + W1_KT*W1_NT*FRAG_HALVES)
#define W3_OFF (W2_OFF + W2_KT*W2_NT*FRAG_HALVES)
#define W4_OFF (W3_OFF + W3_KT*W3_NT*FRAG_HALVES)
#define WPACK_TOTAL_HALVES (W4_OFF + W4_KT*W4_NT*FRAG_HALVES)

// ---------------------------------------------------------------------------
// CDNA5 hardware tanh (TRANS32 op per ISA flow/sync §8.4) instead of the
// branchy libm tanhf expansion.
// ---------------------------------------------------------------------------
__device__ __forceinline__ float fast_tanh(float x) {
#if __has_builtin(__builtin_amdgcn_tanhf)
    return __builtin_amdgcn_tanhf(x);
#elif __has_builtin(__builtin_amdgcn_tanh_f32)
    return __builtin_amdgcn_tanh_f32(x);
#else
    float y;
    asm volatile("v_tanh_f32 %0, %1" : "=v"(y) : "v"(x));
    return y;
#endif
}

// ---------------------------------------------------------------------------
// Pack fp32 row-major W[K][N] into f16 WMMA B-operand fragments.
// B-matrix 32x16 (KxN), f16, wave32: lane L holds column N = L&15,
// K = (L>=16 ? 16 : 0) + 2*v + h  for VGPR v, half h (contiguous k order chosen here, same as A).
// ---------------------------------------------------------------------------
__global__ void pack_weights_kernel(const float* __restrict__ W,
                                    int Kreal, int Nreal, int Ntiles,
                                    _Float16* __restrict__ dst) {
    const int frag = blockIdx.x;           // kt*Ntiles + nt
    const int lane = threadIdx.x;          // 0..31
    const int kt = frag / Ntiles;
    const int nt = frag - kt * Ntiles;
    const int laneHi = lane >> 4;
    const int n = nt * 16 + (lane & 15);
    _Float16* out = dst + ((size_t)frag * 32 + lane) * 16;
#pragma unroll
    for (int v = 0; v < 8; ++v) {
#pragma unroll
        for (int h = 0; h < 2; ++h) {
            const int k = kt * 32 + laneHi * 16 + v * 2 + h;
            float val = (k < Kreal && n < Nreal) ? W[(size_t)k * Nreal + n] : 0.0f;
            out[v * 2 + h] = (_Float16)val;
        }
    }
}

// A-matrix 16x32 (MxK), f16, wave32 (contiguous k order used here, matching packed B):
// lane L holds row M = L&15; elements 0..15 = K = kt*32 + (L>=16?16:0) + 0..15.
__device__ inline v16h load_a_frag(const _Float16* rowp, int kt, int laneHi) {
    union { v16h v; uint4 u[2]; } r;
    const _Float16* p = rowp + kt * 32 + laneHi * 16;
    r.u[0] = *(const uint4*)(p);
    r.u[1] = *(const uint4*)(p + 8);
    return r.v;
}

__global__ __launch_bounds__(WAVES_PER_WG * 32)
void pinn_forward_kernel(const float* __restrict__ history,
                         const _Float16* __restrict__ wpack,
                         const float* __restrict__ eb1,
                         const float* __restrict__ eb2,
                         const float* __restrict__ cb1,
                         const float* __restrict__ cb2,
                         const float* __restrict__ alpha_p,
                         const float* __restrict__ beta_p,
                         const float* __restrict__ gamma_p,
                         const float* __restrict__ lam_p,
                         float* __restrict__ out) {
    __shared__ alignas(16) _Float16 act[WAVES_PER_WG][16][HID];
    __shared__ float phys[WAVES_PER_WG][16][FORE];

    const int tid    = threadIdx.x;
    const int w      = tid >> 5;
    const int lane   = tid & 31;
    const int laneHi = lane >> 4;
    const int laneN  = lane & 15;
    const int mBase  = laneHi * 8;
    const int row0   = (blockIdx.x * WAVES_PER_WG + w) * ROWS_PER_WAVE;

    const float aC  = 1.0f / (1.0f + __expf(-alpha_p[0]));
    const float bC  = 1.0f / (1.0f + __expf(-beta_p[0]));
    const float gC  = fabsf(gamma_p[0]);
    const float lam = 1.0f / (1.0f + __expf(-lam_p[0]));

    float* outPred = out;
    float* outPhys = out + (size_t)B_ROWS * FORE;
    float* outSoft = out + 2 * (size_t)B_ROWS * FORE;

    // ---------------- Phase 1: physics recurrence (lanes 0..15) -------------
    if (lane < 16) {
        const int row = row0 + lane;
        const float* hrow = history + (size_t)row * HIST;
        float hv[HIST];
#pragma unroll
        for (int j = 0; j < HIST; ++j) hv[j] = hrow[j];

        _Float16* crow = &act[w][lane][0];
#pragma unroll
        for (int j = 0; j < HIST; ++j) crow[j] = (_Float16)hv[j];

        float buf[TAU_INT];
#pragma unroll
        for (int j = 0; j < TAU_INT; ++j) buf[j] = hv[HIST - TAU_INT + j];
        float T = hv[HIST - 1];

        int idx = 0;
#pragma unroll
        for (int s = 0; s < FORE; ++s) {
            const float Td = buf[idx];
            const float Tn = T - aC * T - bC * Td - gC * T * T * T;
            buf[idx] = Tn;                 // pop-front + push-back == ring
            idx = (idx + 1) % TAU_INT;     // constant-folds under full unroll
            T = Tn;
            crow[HIST + s] = (_Float16)Tn;
            phys[w][lane][s] = Tn;
            outPhys[(size_t)row * FORE + s] = Tn;
        }
#pragma unroll
        for (int j = HIST + FORE; j < 64; ++j) crow[j] = (_Float16)0.0f;
    }
    __syncthreads();

    const _Float16* aRow = &act[w][laneN][0];

    // ---------------- Layer 1: [16 x 64] @ [64 x 256], tanh -----------------
    {
        const v16h a0 = load_a_frag(aRow, 0, laneHi);
        const v16h a1 = load_a_frag(aRow, 1, laneHi);
        __syncthreads();  // A in regs; LDS buffer now reusable for outputs
        const _Float16* wp = wpack + W1_OFF;
#pragma unroll 1
        for (int nt = 0; nt < 16; ++nt) {
            const float bv = eb1[nt * 16 + laneN];
            v8f c = {bv, bv, bv, bv, bv, bv, bv, bv};
            v16h b0 = *(const v16h*)(wp + ((size_t)(0 * W1_NT + nt) * 32 + lane) * 16);
            c = __builtin_amdgcn_wmma_f32_16x16x32_f16(false, a0, false, b0,
                                                       (short)0, c, false, false);
            v16h b1 = *(const v16h*)(wp + ((size_t)(1 * W1_NT + nt) * 32 + lane) * 16);
            c = __builtin_amdgcn_wmma_f32_16x16x32_f16(false, a1, false, b1,
                                                       (short)0, c, false, false);
#pragma unroll
            for (int r = 0; r < 8; ++r)
                act[w][mBase + r][nt * 16 + laneN] = (_Float16)fast_tanh(c[r]);
        }
    }
    __syncthreads();

    // ------------- Layers 2 & 3: [16 x 256] @ [256 x 256], tanh -------------
#pragma unroll 1
    for (int layer = 0; layer < 2; ++layer) {
        const _Float16* wp  = wpack + (layer == 0 ? W2_OFF : W3_OFF);
        const float*   bias = (layer == 0) ? eb2 : cb1;

        v16h af[8];
#pragma unroll
        for (int kt = 0; kt < 8; ++kt) af[kt] = load_a_frag(aRow, kt, laneHi);
        __syncthreads();  // fence: A fully in regs before LDS overwrite

#pragma unroll 1
        for (int nt = 0; nt < 16; ++nt) {
            const float bv = bias[nt * 16 + laneN];
            v8f c = {bv, bv, bv, bv, bv, bv, bv, bv};
#pragma unroll
            for (int kt = 0; kt < 8; ++kt) {
                v16h bf = *(const v16h*)(wp + ((size_t)(kt * 16 + nt) * 32 + lane) * 16);
                c = __builtin_amdgcn_wmma_f32_16x16x32_f16(false, af[kt], false, bf,
                                                           (short)0, c, false, false);
            }
#pragma unroll
            for (int r = 0; r < 8; ++r)
                act[w][mBase + r][nt * 16 + laneN] = (_Float16)fast_tanh(c[r]);
        }
        __syncthreads();
    }

    // ------------- Layer 4: [16 x 256] @ [256 x 32(real 20)] + mix ----------
    {
        v16h af[8];
#pragma unroll
        for (int kt = 0; kt < 8; ++kt) af[kt] = load_a_frag(aRow, kt, laneHi);
        const _Float16* wp = wpack + W4_OFF;
#pragma unroll 1
        for (int nt = 0; nt < 2; ++nt) {
            const int col = nt * 16 + laneN;
            const float bv = (col < FORE) ? cb2[col] : 0.0f;
            v8f c = {bv, bv, bv, bv, bv, bv, bv, bv};
#pragma unroll
            for (int kt = 0; kt < 8; ++kt) {
                v16h bf = *(const v16h*)(wp + ((size_t)(kt * W4_NT + nt) * 32 + lane) * 16);
                c = __builtin_amdgcn_wmma_f32_16x16x32_f16(false, af[kt], false, bf,
                                                           (short)0, c, false, false);
            }
            if (col < FORE) {
#pragma unroll
                for (int r = 0; r < 8; ++r) {
                    const int m   = mBase + r;
                    const int row = row0 + m;
                    const float soft = c[r];
                    const float p    = phys[w][m][col];
                    outSoft[(size_t)row * FORE + col] = soft;
                    outPred[(size_t)row * FORE + col] = p + lam * soft;
                }
            }
        }
    }
}

extern "C" void kernel_launch(void* const* d_in, const int* in_sizes, int n_in,
                              void* d_out, int out_size, void* d_ws, size_t ws_size,
                              hipStream_t stream) {
    const float* history = (const float*)d_in[0];
    const float* enc_w1  = (const float*)d_in[1];
    const float* enc_b1  = (const float*)d_in[2];
    const float* enc_w2  = (const float*)d_in[3];
    const float* enc_b2  = (const float*)d_in[4];
    const float* cor_w1  = (const float*)d_in[5];
    const float* cor_b1  = (const float*)d_in[6];
    const float* cor_w2  = (const float*)d_in[7];
    const float* cor_b2  = (const float*)d_in[8];
    const float* alpha   = (const float*)d_in[9];
    const float* beta    = (const float*)d_in[10];
    const float* gamma   = (const float*)d_in[11];
    // d_in[12] = tau (unused by reference math)
    const float* lambda_mix = (const float*)d_in[13];

    _Float16* wpack = (_Float16*)d_ws;
    float* out = (float*)d_out;

    // One-time (per launch) weight repack into WMMA B-fragment order.
    pack_weights_kernel<<<W1_KT * W1_NT, 32, 0, stream>>>(enc_w1, HIST + FORE, HID, W1_NT, wpack + W1_OFF);
    pack_weights_kernel<<<W2_KT * W2_NT, 32, 0, stream>>>(enc_w2, HID, HID, W2_NT, wpack + W2_OFF);
    pack_weights_kernel<<<W3_KT * W3_NT, 32, 0, stream>>>(cor_w1, HID, HID, W3_NT, wpack + W3_OFF);
    pack_weights_kernel<<<W4_KT * W4_NT, 32, 0, stream>>>(cor_w2, HID, FORE, W4_NT, wpack + W4_OFF);

    const int grid = B_ROWS / ROWS_PER_WG;  // 4096
    pinn_forward_kernel<<<grid, WAVES_PER_WG * 32, 0, stream>>>(
        history, wpack, enc_b1, enc_b2, cor_b1, cor_b2,
        alpha, beta, gamma, lambda_mix, out);
}
